// GraphRec_45440753992066
// MI455X (gfx1250) — matrix operand order repaired and checked by reading, not compiled, not run on hardware
//
#include <hip/hip_runtime.h>
#include <hip/hip_bf16.h>
#include <math.h>

typedef _Float16 v16h __attribute__((ext_vector_type(16)));
typedef _Float16 h8   __attribute__((ext_vector_type(8)));
typedef float    v8f  __attribute__((ext_vector_type(8)));

#define B_   2048
#define LU_  50
#define LS_  32
#define LV_  50
#define EPS_ 1e-5f

// ---------------------------------------------------------------------------
// WMMA GEMM: Y[M,N] = act( X[M,K] @ W[K,N] + bias )
// K = KT*32, N = NT*16.  X rows are assembled in LDS from one or two 64-wide
// f32 sources: half0 = src0[(idx0?idx0[i]:i)*64 + k], half1 (if K==128) =
// src1[(idx1?idx1[i/div1]:i)*64 + k-64].  This fuses embedding gather,
// broadcast-center concat, and dense concat into the GEMM staging.
// Block = 128 threads = 4 waves; each wave computes one 16-row M-tile x N.
// ---------------------------------------------------------------------------
template<int KT, int NT, bool RELU>
__global__ __launch_bounds__(128)
void gemm_k(int M,
            const float* __restrict__ src0, const int* __restrict__ idx0,
            const float* __restrict__ src1, const int* __restrict__ idx1, int div1,
            const float* __restrict__ W, const float* __restrict__ bias,
            float* __restrict__ Y)
{
    constexpr int K = KT * 32;
    constexpr int N = NT * 16;
    __shared__ _Float16 sX[64 * K];   // 64 rows, row-major
    __shared__ _Float16 sW[N * K];    // W transposed: [n][k]

    const int tid    = threadIdx.x;
    const int block0 = blockIdx.x * 64;

    // stage W (f32 -> f16, transposed so each output column is a K-run)
    for (int i = tid; i < N * K; i += 128) {
        int n = i / K, k = i % K;
        sW[n * K + k] = (_Float16)W[k * N + n];
    }
    // stage X tile, assembling the concat / gather on the fly
    for (int i = tid; i < 64 * K; i += 128) {
        int r = i / K, k = i % K;
        int gr = block0 + r;
        float v = 0.f;
        if (gr < M) {
            if (k < 64) {
                int row = idx0 ? idx0[gr] : gr;
                v = src0[row * 64 + k];
            } else {
                int row = idx1 ? idx1[gr / div1] : gr;
                v = src1[row * 64 + (k - 64)];
            }
        }
        sX[r * K + k] = (_Float16)v;
    }
    __syncthreads();

    const int  wave = tid >> 5;
    const int  lane = tid & 31;
    const int  l16  = lane & 15;
    const bool hi   = lane >= 16;

    v8f zero = {};
    v8f acc[NT];
#pragma unroll
    for (int nt = 0; nt < NT; ++nt) acc[nt] = zero;

#pragma unroll
    for (int kk = 0; kk < KT; ++kk) {
        // A fragment: lane<16 holds K {0..7,16..23}; lane>=16 holds {8..15,24..31}
        const _Float16* ar = &sX[(wave * 16 + l16) * K + kk * 32];
        const int aoff = hi ? 8 : 0;
        h8 a0 = *(const h8*)(ar + aoff);
        h8 a1 = *(const h8*)(ar + aoff + 16);
        v16h af = __builtin_shufflevector(a0, a1, 0,1,2,3,4,5,6,7,8,9,10,11,12,13,14,15);
#pragma unroll
        for (int nt = 0; nt < NT; ++nt) {
            // B fragment: column n = l16, K {0..7,16..23} / {8..15,24..31}
            const _Float16* br = &sW[(nt * 16 + l16) * K + kk * 32];
            const int boff = hi ? 8 : 0;
            h8 b0 = *(const h8*)(br + boff);
            h8 b1 = *(const h8*)(br + boff + 16);
            v16h bf = __builtin_shufflevector(b0, b1, 0,1,2,3,4,5,6,7,8,9,10,11,12,13,14,15);
            acc[nt] = __builtin_amdgcn_wmma_f32_16x16x32_f16(
                false, af, false, bf, (short)0, acc[nt], false, false);
        }
    }

    // D layout: VGPR r -> M=r (lanes 0-15) / M=8+r (lanes 16-31), N = l16
    const int mbase = block0 + wave * 16 + (hi ? 8 : 0);
#pragma unroll
    for (int r = 0; r < 8; ++r) {
        int gr = mbase + r;
        if (gr < M) {
#pragma unroll
            for (int nt = 0; nt < NT; ++nt) {
                int   n = nt * 16 + l16;
                float v = acc[nt][r] + bias[n];
                if (RELU) v = v > 0.f ? v : 0.f;
                Y[gr * N + n] = v;
            }
        }
    }
}

// ---------------------------------------------------------------------------
// Attention tail: s_l = h2[b,l,:].W3 + b3 ; alpha = softmax_L(s) ;
// out[b,:] = sum_l alpha_l * V[b,l,:]   (V dense from ws, or gathered)
// One block (64 threads) per batch row; L <= 64.
// ---------------------------------------------------------------------------
__global__ __launch_bounds__(64)
void attn_k(int L,
            const float* __restrict__ h2,     // [B*L, 64]
            const float* __restrict__ W3,     // [64]
            const float* __restrict__ b3,     // [1]
            const float* __restrict__ vdense, // [B*L,64] or null
            const float* __restrict__ vtable, // embedding table or null
            const int*   __restrict__ vidx,   // [B*L] or null
            float* __restrict__ out)          // [B, 64]
{
    __shared__ float s[64];
    __shared__ float w3s[64];
    const int b = blockIdx.x;
    const int t = threadIdx.x;
    w3s[t] = W3[t];
    __syncthreads();
    float sv = -1e30f;
    if (t < L) {
        const float* row = h2 + (size_t)(b * L + t) * 64;
        float a = b3[0];
        for (int d = 0; d < 64; ++d) a += row[d] * w3s[d];
        sv = a;
    }
    s[t] = sv;
    __syncthreads();
    float mx = -1e30f;
    for (int l = 0; l < L; ++l) mx = fmaxf(mx, s[l]);
    float sum = 0.f;
    for (int l = 0; l < L; ++l) sum += __expf(s[l] - mx);
    const float inv = 1.f / sum;
    float acc = 0.f;
    for (int l = 0; l < L; ++l) {
        float a = __expf(s[l] - mx) * inv;
        const float* vr = vidx ? (vtable + (size_t)vidx[b * L + l] * 64)
                               : (vdense + (size_t)(b * L + l) * 64);
        acc += a * vr[t];
    }
    out[(size_t)b * 64 + t] = acc;
}

// ---------------------------------------------------------------------------
// BatchNorm (biased variance, as jnp.var) + ReLU, in place. One block/column.
// ---------------------------------------------------------------------------
__global__ __launch_bounds__(256)
void bn_relu_k(int M, int N, float* __restrict__ X,
               const float* __restrict__ g, const float* __restrict__ bb)
{
    __shared__ float r1[256], r2[256];
    const int c = blockIdx.x;
    const int t = threadIdx.x;
    float s = 0.f, ss = 0.f;
    for (int r = t; r < M; r += 256) {
        float v = X[(size_t)r * N + c];
        s += v; ss += v * v;
    }
    r1[t] = s; r2[t] = ss;
    __syncthreads();
    for (int o = 128; o > 0; o >>= 1) {
        if (t < o) { r1[t] += r1[t + o]; r2[t] += r2[t + o]; }
        __syncthreads();
    }
    const float mean = r1[0] / (float)M;
    const float var  = r2[0] / (float)M - mean * mean;
    const float iv   = rsqrtf(var + EPS_);
    const float gg = g[c], bc = bb[c];
    for (int r = t; r < M; r += 256) {
        float v = X[(size_t)r * N + c];
        v = gg * (v - mean) * iv + bc;
        X[(size_t)r * N + c] = v > 0.f ? v : 0.f;
    }
}

// scores[b] = x16[b,:] . W + bias
__global__ void final_k(int M, const float* __restrict__ X16,
                        const float* __restrict__ W, const float* __restrict__ b,
                        float* __restrict__ out)
{
    int i = blockIdx.x * blockDim.x + threadIdx.x;
    if (i < M) {
        float acc = b[0];
        for (int j = 0; j < 16; ++j) acc += X16[i * 16 + j] * W[j];
        out[i] = acc;
    }
}

// ---------------------------------------------------------------------------

#define GEMM(KT, NT, RELU, M, s0, i0, s1, i1, dv, Wp, bp, Yp)                 \
    gemm_k<KT, NT, RELU><<<dim3(((M) + 63) / 64), dim3(128), 0, stream>>>(    \
        (M), (s0), (i0), (s1), (i1), (dv), (Wp), (bp), (Yp))

extern "C" void kernel_launch(void* const* d_in, const int* in_sizes, int n_in,
                              void* d_out, int out_size, void* d_ws, size_t ws_size,
                              hipStream_t stream)
{
    (void)in_sizes; (void)n_in; (void)out_size; (void)ws_size;
    // --- inputs (setup_inputs dict order; params dict flattened in order) ---
    const int* nodes_u        = (const int*)d_in[0];
    const int* nodes_v        = (const int*)d_in[1];
    const int* u_hist_items   = (const int*)d_in[2];
    const int* u_hist_ratings = (const int*)d_in[3];
    const int* u_social       = (const int*)d_in[4];
    const int* v_hist_users   = (const int*)d_in[5];
    const int* v_hist_ratings = (const int*)d_in[6];
    const float* u2e = (const float*)d_in[7];
    const float* v2e = (const float*)d_in[8];
    const float* r2e = (const float*)d_in[9];
    const float *gv_W1=(const float*)d_in[10],  *gv_b1=(const float*)d_in[11],
                *gv_W2=(const float*)d_in[12],  *gv_b2=(const float*)d_in[13],
                *gv_W3=(const float*)d_in[14],  *gv_b3=(const float*)d_in[15];
    const float *attI_W1=(const float*)d_in[16],*attI_b1=(const float*)d_in[17],
                *attI_W2=(const float*)d_in[18],*attI_b2=(const float*)d_in[19],
                *attI_W3=(const float*)d_in[20],*attI_b3=(const float*)d_in[21];
    const float *attS_W1=(const float*)d_in[22],*attS_b1=(const float*)d_in[23],
                *attS_W2=(const float*)d_in[24],*attS_b2=(const float*)d_in[25],
                *attS_W3=(const float*)d_in[26],*attS_b3=(const float*)d_in[27];
    const float *w_r1_W=(const float*)d_in[28], *w_r1_b=(const float*)d_in[29],
                *w_r2_W=(const float*)d_in[30], *w_r2_b=(const float*)d_in[31];
    const float *umlp_W1=(const float*)d_in[32],*umlp_b1=(const float*)d_in[33],
                *umlp_W2=(const float*)d_in[34],*umlp_b2=(const float*)d_in[35],
                *umlp_W3=(const float*)d_in[36],*umlp_b3=(const float*)d_in[37];
    const float *gu_W1=(const float*)d_in[38],  *gu_b1=(const float*)d_in[39],
                *gu_W2=(const float*)d_in[40],  *gu_b2=(const float*)d_in[41],
                *gu_W3=(const float*)d_in[42],  *gu_b3=(const float*)d_in[43];
    const float *attV_W1=(const float*)d_in[44],*attV_b1=(const float*)d_in[45],
                *attV_W2=(const float*)d_in[46],*attV_b2=(const float*)d_in[47],
                *attV_W3=(const float*)d_in[48],*attV_b3=(const float*)d_in[49];
    const float *vw_r1_W=(const float*)d_in[50],*vw_r1_b=(const float*)d_in[51],
                *vw_r2_W=(const float*)d_in[52],*vw_r2_b=(const float*)d_in[53];
    const float *w_ur1_W=(const float*)d_in[54],*w_ur1_b=(const float*)d_in[55],
                *w_ur2_W=(const float*)d_in[56],*w_ur2_b=(const float*)d_in[57];
    const float *w_vr1_W=(const float*)d_in[58],*w_vr1_b=(const float*)d_in[59],
                *w_vr2_W=(const float*)d_in[60],*w_vr2_b=(const float*)d_in[61];
    const float *w_uv1_W=(const float*)d_in[62],*w_uv1_b=(const float*)d_in[63],
                *w_uv2_W=(const float*)d_in[64],*w_uv2_b=(const float*)d_in[65],
                *w_uv3_W=(const float*)d_in[66],*w_uv3_b=(const float*)d_in[67];
    const float *bn1_g=(const float*)d_in[68], *bn1_b=(const float*)d_in[69],
                *bn2_g=(const float*)d_in[70], *bn2_b=(const float*)d_in[71],
                *bn3_g=(const float*)d_in[72], *bn3_b=(const float*)d_in[73],
                *bn4_g=(const float*)d_in[74], *bn4_b=(const float*)d_in[75];

    const int MU = B_ * LU_, MS = B_ * LS_, MV = B_ * LV_;
    const int* NOIDX = nullptr;
    const float* NOSRC = nullptr;

    // --- workspace layout (floats) ---
    float* ws = (float*)d_ws;
    float* H1 = ws;                        // [MU,64]
    float* H2 = H1 + (size_t)MU * 64;      // [MU,64]
    float* XB = H2 + (size_t)MU * 64;      // xia / fjt [MU,64]
    float* sm = XB + (size_t)MU * 64;
    float* hiI_raw = sm + 0 * (size_t)B_ * 64;
    float* hiS     = sm + 1 * (size_t)B_ * 64;
    float* tmpA    = sm + 2 * (size_t)B_ * 64;
    float* hiI     = sm + 3 * (size_t)B_ * 64;
    float* hi      = sm + 4 * (size_t)B_ * 64;
    float* zj_raw  = sm + 5 * (size_t)B_ * 64;
    float* zj      = sm + 6 * (size_t)B_ * 64;
    float* tA      = sm + 7 * (size_t)B_ * 64;
    float* tB      = sm + 8 * (size_t)B_ * 64;
    float* x16b    = sm + 9 * (size_t)B_ * 64;

    // ============================ user modeling ============================
    // xia = mlp3(concat(v2e[u_hist_items], r2e[u_hist_ratings]), gv)
    GEMM(4,4,true,  MU, v2e, u_hist_items, r2e, u_hist_ratings, 1, gv_W1, gv_b1, H1);
    GEMM(2,4,true,  MU, H1, NOIDX, NOSRC, NOIDX, 1, gv_W2, gv_b2, H2);
    GEMM(2,4,false, MU, H2, NOIDX, NOSRC, NOIDX, 1, gv_W3, gv_b3, XB);
    // attI: mlp3(concat(xia, pi_broadcast)) -> softmax -> hi_I_raw
    GEMM(4,4,true,  MU, XB, NOIDX, u2e, nodes_u, LU_, attI_W1, attI_b1, H1);
    GEMM(2,4,true,  MU, H1, NOIDX, NOSRC, NOIDX, 1, attI_W2, attI_b2, H2);
    attn_k<<<dim3(B_), dim3(64), 0, stream>>>(LU_, H2, attI_W3, attI_b3,
                                              XB, NOSRC, NOIDX, hiI_raw);
    // attS over social neighbours (values gathered from u2e)
    GEMM(4,4,true,  MS, u2e, u_social, u2e, nodes_u, LS_, attS_W1, attS_b1, H1);
    GEMM(2,4,true,  MS, H1, NOIDX, NOSRC, NOIDX, 1, attS_W2, attS_b2, H2);
    attn_k<<<dim3(B_), dim3(64), 0, stream>>>(LS_, H2, attS_W3, attS_b3,
                                              NOSRC, u2e, u_social, hiS);
    // hi_I = relu(relu(hi_I_raw @ w_r1) @ w_r2)
    GEMM(2,4,true,  B_, hiI_raw, NOIDX, NOSRC, NOIDX, 1, w_r1_W, w_r1_b, tmpA);
    GEMM(2,4,true,  B_, tmpA, NOIDX, NOSRC, NOIDX, 1, w_r2_W, w_r2_b, hiI);
    // hi = mlp3(concat(hi_I, hi_S), umlp)
    GEMM(4,4,true,  B_, hiI, NOIDX, hiS, NOIDX, 1, umlp_W1, umlp_b1, tA);
    GEMM(2,4,true,  B_, tA, NOIDX, NOSRC, NOIDX, 1, umlp_W2, umlp_b2, tB);
    GEMM(2,4,false, B_, tB, NOIDX, NOSRC, NOIDX, 1, umlp_W3, umlp_b3, hi);

    // ============================ item modeling ============================
    GEMM(4,4,true,  MV, u2e, v_hist_users, r2e, v_hist_ratings, 1, gu_W1, gu_b1, H1);
    GEMM(2,4,true,  MV, H1, NOIDX, NOSRC, NOIDX, 1, gu_W2, gu_b2, H2);
    GEMM(2,4,false, MV, H2, NOIDX, NOSRC, NOIDX, 1, gu_W3, gu_b3, XB);   // fjt
    GEMM(4,4,true,  MV, XB, NOIDX, v2e, nodes_v, LV_, attV_W1, attV_b1, H1);
    GEMM(2,4,true,  MV, H1, NOIDX, NOSRC, NOIDX, 1, attV_W2, attV_b2, H2);
    attn_k<<<dim3(B_), dim3(64), 0, stream>>>(LV_, H2, attV_W3, attV_b3,
                                              XB, NOSRC, NOIDX, zj_raw);
    GEMM(2,4,true,  B_, zj_raw, NOIDX, NOSRC, NOIDX, 1, vw_r1_W, vw_r1_b, tA);
    GEMM(2,4,true,  B_, tA, NOIDX, NOSRC, NOIDX, 1, vw_r2_W, vw_r2_b, zj);

    // ============================ rating head ==============================
    GEMM(2,4,false, B_, hi, NOIDX, NOSRC, NOIDX, 1, w_ur1_W, w_ur1_b, tA);
    bn_relu_k<<<dim3(64), dim3(256), 0, stream>>>(B_, 64, tA, bn1_g, bn1_b);
    GEMM(2,4,false, B_, tA, NOIDX, NOSRC, NOIDX, 1, w_ur2_W, w_ur2_b, hiI);  // hi'
    GEMM(2,4,false, B_, zj, NOIDX, NOSRC, NOIDX, 1, w_vr1_W, w_vr1_b, tB);
    bn_relu_k<<<dim3(64), dim3(256), 0, stream>>>(B_, 64, tB, bn2_g, bn2_b);
    GEMM(2,4,false, B_, tB, NOIDX, NOSRC, NOIDX, 1, w_vr2_W, w_vr2_b, hiS);  // zj'
    GEMM(4,4,false, B_, hiI, NOIDX, hiS, NOIDX, 1, w_uv1_W, w_uv1_b, tmpA);
    bn_relu_k<<<dim3(64), dim3(256), 0, stream>>>(B_, 64, tmpA, bn3_g, bn3_b);
    GEMM(2,1,false, B_, tmpA, NOIDX, NOSRC, NOIDX, 1, w_uv2_W, w_uv2_b, x16b);
    bn_relu_k<<<dim3(16), dim3(256), 0, stream>>>(B_, 16, x16b, bn4_g, bn4_b);
    final_k<<<dim3((B_ + 255) / 256), dim3(256), 0, stream>>>(
        B_, x16b, w_uv3_W, w_uv3_b, (float*)d_out);
}
